// TargetAttention_30356828848807
// MI455X (gfx1250) — matrix operand order repaired and checked by reading, not compiled, not run on hardware
//
#include <hip/hip_runtime.h>

typedef __attribute__((ext_vector_type(16))) __bf16 v16bf;
typedef __attribute__((ext_vector_type(8)))  float  v8f;

#define B_    4
#define S_    4096
#define D_    512
#define L_    8921
#define LT64_ 140                       // ceil(L/64)
#define LT32_ 279                       // ceil(L/32)
#define LT16_ 558                       // ceil(L/16)
#define SCALE_ 0.044194173824159216f    // 1/sqrt(512)

#define QP_  (D_ + 8)                   // Qlds row stride (bf16): +16B -> bank spread
#define AP_  (S_ + 8)                   // Alds row stride (bf16)
#define EP_  (S_ + 16)                  // Elds row stride (f32)

// ---------------------------------------------------------------------------
// One-time fixups into d_ws (~33.5MB, L2-resident afterwards).
// ---------------------------------------------------------------------------
__global__ __launch_bounds__(256) void convert_f32_bf16_kernel(
    const float* __restrict__ src, __bf16* __restrict__ dst, int n)
{
    int i = (blockIdx.x * 256 + threadIdx.x) * 4;
    if (i + 3 < n) {
        float4 v = *(const float4*)(src + i);
        dst[i + 0] = (__bf16)v.x;  dst[i + 1] = (__bf16)v.y;
        dst[i + 2] = (__bf16)v.z;  dst[i + 3] = (__bf16)v.w;
    }
}

// V [B,S,D] f32  ->  Vt [B,D,S] bf16
__global__ __launch_bounds__(256) void transpose_v_kernel(
    const float* __restrict__ V, __bf16* __restrict__ Vt)
{
    __shared__ __bf16 tile[32][33];
    const int bb = blockIdx.z;
    const int s0 = blockIdx.x * 32;
    const int d0 = blockIdx.y * 32;
    const int tx = threadIdx.x, ty = threadIdx.y;
    for (int i = ty; i < 32; i += 8)
        tile[i][tx] = (__bf16)V[((size_t)bb * S_ + s0 + i) * D_ + d0 + tx];
    __syncthreads();
    for (int i = ty; i < 32; i += 8)
        Vt[((size_t)bb * D_ + d0 + i) * S_ + s0 + tx] = tile[tx][i];
}

// ---------------------------------------------------------------------------
// B-fragment loaders (contiguous bf16 fast path / f32+cvt fallback).
// rowBase is an element offset to the start of the source row.
// ---------------------------------------------------------------------------
template <typename T>
__device__ __forceinline__ v16bf load_bfrag_row(const T* p, size_t rowBase,
                                                int kk, int half)
{
    if constexpr (sizeof(T) == 2) {
        return *(const v16bf*)((const __bf16*)p + rowBase + kk * 32 + half * 16);
    } else {
        const float* r = (const float*)p + rowBase + kk * 32 + half * 16;
        v16bf b;
        #pragma unroll
        for (int j = 0; j < 16; ++j) b[j] = (__bf16)r[j];
        return b;
    }
}

// strided fallback for V [B,S,D] (column d fixed, k walks rows)
__device__ __forceinline__ v16bf load_bfrag_strided(const float* p, size_t base,
                                                    int kk, int half)
{
    const float* r = p + base + (size_t)(kk * 32 + half * 16) * D_;
    v16bf b;
    #pragma unroll
    for (int j = 0; j < 16; ++j) b[j] = (__bf16)r[(size_t)j * D_];
    return b;
}

// ---------------------------------------------------------------------------
// Kernel A: raw logits E = (Q @ K^T) * scale -> written to A region (scratch).
// 64 L-rows per workgroup; each B-fragment reused across 4 row-tiles
// (256B of L2 traffic per WMMA). B-frag software-pipelined one step ahead;
// A-frags batch-loaded so the 4 WMMAs issue back-to-back.
// ---------------------------------------------------------------------------
template <typename TK>
__global__ __launch_bounds__(256) void qk_gemm_kernel(
    const float* __restrict__ Q, const TK* __restrict__ Km,
    float* __restrict__ E)
{
    __shared__ __bf16 Qlds[64 * QP_];   // ~65 KB

    const int lt = blockIdx.x;
    const int bb = blockIdx.y;
    const int t  = threadIdx.x;
    const int l0 = lt * 64;

    for (int e = t * 4; e < 64 * D_; e += 1024) {
        int r = e >> 9, c = e & (D_ - 1);
        int l = l0 + r; if (l >= L_) l = L_ - 1;
        float4 q = *(const float4*)&Q[((size_t)bb * L_ + l) * D_ + c];
        __bf16* dst = Qlds + r * QP_ + c;
        dst[0] = (__bf16)q.x; dst[1] = (__bf16)q.y;
        dst[2] = (__bf16)q.z; dst[3] = (__bf16)q.w;
    }
    __syncthreads();

    const int wave = t >> 5;
    const int lane = t & 31;
    const int half = lane >> 4;
    const int lm   = lane & 15;

    for (int tc = wave; tc < S_ / 16; tc += 8) {
        const int s0 = tc * 16;
        if (tc + 8 < S_ / 16)
            __builtin_prefetch(Km + ((size_t)bb * S_ + s0 + 128 + lm) * D_, 0, 0);

        const size_t rb = ((size_t)bb * S_ + (s0 + lm)) * (size_t)D_;
        v8f acc[4] = {};
        v16bf bnext = load_bfrag_row(Km, rb, 0, half);
        #pragma unroll 4
        for (int kk = 0; kk < D_ / 32; ++kk) {
            v16bf b = bnext;
            if (kk + 1 < D_ / 32)
                bnext = load_bfrag_row(Km, rb, kk + 1, half);
            v16bf a[4];
            #pragma unroll
            for (int rt = 0; rt < 4; ++rt) {
                const __bf16* qa = Qlds + (rt * 16 + lm) * QP_
                                   + kk * 32 + half * 16;
                #pragma unroll
                for (int j = 0; j < 8; ++j) {
                    a[rt][j] = qa[j]; a[rt][8 + j] = qa[8 + j];
                }
            }
            #pragma unroll
            for (int rt = 0; rt < 4; ++rt)
                acc[rt] = __builtin_amdgcn_wmma_f32_16x16x32_bf16(
                              false, a[rt], false, b, (short)0, acc[rt],
                              false, false);
        }
        #pragma unroll
        for (int rt = 0; rt < 4; ++rt) {
            #pragma unroll
            for (int i = 0; i < 8; ++i) {
                int l = l0 + rt * 16 + i + half * 8;
                if (l < L_)
                    E[((size_t)bb * L_ + l) * S_ + s0 + lm] = acc[rt][i] * SCALE_;
            }
        }
    }
}

// ---------------------------------------------------------------------------
// Kernel B: in-place row softmax of A ([B,L,S], rows hold scaled logits).
// 16 rows per workgroup, 16 threads per row, float4, barrier-free.
// ---------------------------------------------------------------------------
__global__ __launch_bounds__(256) void softmax_kernel(float* __restrict__ A)
{
    __shared__ float Elds[16 * EP_];    // ~257 KB, padded stride

    const int lt = blockIdx.x;
    const int bb = blockIdx.y;
    const int t  = threadIdx.x;
    const int l0 = lt * 16;
    const int r  = t >> 4;
    const int c0 = t & 15;

    int lrow = l0 + r; if (lrow >= L_) lrow = L_ - 1;
    float* row = A + ((size_t)bb * L_ + lrow) * S_;

    float m = -3.0e38f;
    for (int c4 = c0; c4 < S_ / 4; c4 += 16) {
        float4 v = *(const float4*)&row[c4 * 4];
        *(float4*)&Elds[r * EP_ + c4 * 4] = v;
        m = fmaxf(m, fmaxf(fmaxf(v.x, v.y), fmaxf(v.z, v.w)));
    }
    #pragma unroll
    for (int i = 1; i < 16; i <<= 1) m = fmaxf(m, __shfl_xor(m, i, 16));

    float sum = 0.0f;
    for (int c4 = c0; c4 < S_ / 4; c4 += 16) {
        float4 v = *(float4*)&Elds[r * EP_ + c4 * 4];
        v.x = __expf(v.x - m); v.y = __expf(v.y - m);
        v.z = __expf(v.z - m); v.w = __expf(v.w - m);
        *(float4*)&Elds[r * EP_ + c4 * 4] = v;
        sum += (v.x + v.y) + (v.z + v.w);
    }
    #pragma unroll
    for (int i = 1; i < 16; i <<= 1) sum += __shfl_xor(sum, i, 16);
    const float inv = 1.0f / sum;

    if (l0 + r < L_) {
        for (int c4 = c0; c4 < S_ / 4; c4 += 16) {
            float4 v = *(float4*)&Elds[r * EP_ + c4 * 4];
            v.x *= inv; v.y *= inv; v.z *= inv; v.w *= inv;
            *(float4*)&row[c4 * 4] = v;
        }
    }
}

// ---------------------------------------------------------------------------
// Kernel C: C = A @ V. 32 L-rows per workgroup (A panel ~256KB LDS, padded);
// each B-fragment reused across 2 row-tiles, pipelined one kk-step ahead.
// TV = __bf16 -> transposed Vt [B,D,S], contiguous loads; float -> fallback.
// ---------------------------------------------------------------------------
template <typename TV>
__global__ __launch_bounds__(256) void av_gemm_kernel(
    const float* __restrict__ A, const TV* __restrict__ Vsrc,
    float* __restrict__ C)
{
    __shared__ __bf16 Alds[32 * AP_];   // ~256.5 KB

    const int lt = blockIdx.x;
    const int bb = blockIdx.y;
    const int t  = threadIdx.x;
    const int l0 = lt * 32;

    for (int e = t * 4; e < 32 * S_; e += 1024) {
        int r = e >> 12, c = e & (S_ - 1);
        int l = l0 + r; if (l >= L_) l = L_ - 1;
        float4 a4 = *(const float4*)&A[((size_t)bb * L_ + l) * S_ + c];
        __bf16* dst = Alds + r * AP_ + c;
        dst[0] = (__bf16)a4.x; dst[1] = (__bf16)a4.y;
        dst[2] = (__bf16)a4.z; dst[3] = (__bf16)a4.w;
    }
    __syncthreads();

    const int wave = t >> 5;
    const int lane = t & 31;
    const int half = lane >> 4;
    const int lm   = lane & 15;

    for (int dt = wave; dt < D_ / 16; dt += 8) {
        const int d0 = dt * 16;
        v8f acc[2] = {};

        v16bf bnext;
        size_t rb = 0;
        if constexpr (sizeof(TV) == 2) {
            rb = ((size_t)bb * D_ + d0 + lm) * (size_t)S_;
            bnext = load_bfrag_row(Vsrc, rb, 0, half);
        } else {
            rb = (size_t)bb * S_ * D_ + d0 + lm;
            bnext = load_bfrag_strided((const float*)Vsrc, rb, 0, half);
        }

        #pragma unroll 4
        for (int kk = 0; kk < S_ / 32; ++kk) {
            v16bf b = bnext;
            if (kk + 1 < S_ / 32) {
                if constexpr (sizeof(TV) == 2)
                    bnext = load_bfrag_row(Vsrc, rb, kk + 1, half);
                else
                    bnext = load_bfrag_strided((const float*)Vsrc, rb, kk + 1, half);
            }
            v16bf a[2];
            #pragma unroll
            for (int rt = 0; rt < 2; ++rt) {
                const __bf16* ar = Alds + (rt * 16 + lm) * AP_
                                   + kk * 32 + half * 16;
                #pragma unroll
                for (int j = 0; j < 8; ++j) {
                    a[rt][j] = ar[j]; a[rt][8 + j] = ar[8 + j];
                }
            }
            #pragma unroll
            for (int rt = 0; rt < 2; ++rt)
                acc[rt] = __builtin_amdgcn_wmma_f32_16x16x32_bf16(
                              false, a[rt], false, b, (short)0, acc[rt],
                              false, false);
        }
        #pragma unroll
        for (int rt = 0; rt < 2; ++rt) {
            #pragma unroll
            for (int i = 0; i < 8; ++i) {
                int l = l0 + rt * 16 + i + half * 8;
                if (l < L_)
                    C[((size_t)bb * L_ + l) * D_ + d0 + lm] = acc[rt][i];
            }
        }
    }
}

// ---------------------------------------------------------------------------
extern "C" void kernel_launch(void* const* d_in, const int* in_sizes, int n_in,
                              void* d_out, int out_size, void* d_ws, size_t ws_size,
                              hipStream_t stream) {
    (void)in_sizes; (void)n_in; (void)out_size;
    const float* Km = (const float*)d_in[0];   // K [B,S,D]
    const float* Vm = (const float*)d_in[1];   // V [B,S,D]
    const float* Qm = (const float*)d_in[2];   // Q [B,L,D]

    float* Cout = (float*)d_out;                               // [B,L,D]
    float* Aout = Cout + (size_t)B_ * L_ * D_;                 // [B,L,S]

    const dim3 g64(LT64_, B_), g16(LT16_, B_), g32(LT32_, B_);
    const size_t kvElems = (size_t)B_ * S_ * D_;
    const size_t needed  = 2 * kvElems * sizeof(__bf16);

    if (ws_size >= needed) {
        __bf16* Kbf = (__bf16*)d_ws;
        __bf16* Vt  = Kbf + kvElems;
        convert_f32_bf16_kernel<<<(int)(kvElems / 4 / 256), 256, 0, stream>>>(
            Km, Kbf, (int)kvElems);
        transpose_v_kernel<<<dim3(S_ / 32, D_ / 32, B_), dim3(32, 8), 0, stream>>>(
            Vm, Vt);
        qk_gemm_kernel<__bf16><<<g64, 256, 0, stream>>>(Qm, Kbf, Aout);
        softmax_kernel<<<g16, 256, 0, stream>>>(Aout);
        av_gemm_kernel<__bf16><<<g32, 256, 0, stream>>>(Aout, Vt, Cout);
    } else {
        qk_gemm_kernel<float><<<g64, 256, 0, stream>>>(Qm, Km, Aout);
        softmax_kernel<<<g16, 256, 0, stream>>>(Aout);
        av_gemm_kernel<float><<<g32, 256, 0, stream>>>(Aout, Vm, Cout);
    }
}
